// StudentNet_6021544149267
// MI455X (gfx1250) — matrix operand order repaired and checked by reading, not compiled, not run on hardware
//
#include <hip/hip_runtime.h>
#include <hip/hip_bf16.h>

typedef __attribute__((ext_vector_type(16))) _Float16 v16h;
typedef __attribute__((ext_vector_type(8)))  float    v8f;

#define WMMA_F32_F16(A,B,C) __builtin_amdgcn_wmma_f32_16x16x32_f16(false,(A),false,(B),(short)0,(C),false,false)

// B=512 N=16 P=16 D=32 E=128 I=256 O=128 NPROT=1024 R=262144 KFIN=65536
#define NPROT 1024
#define RROWS 262144
#define KFIN  65536

__device__ __forceinline__ float rowsum16(float v) {
#pragma unroll
  for (int m = 1; m < 16; m <<= 1) v += __shfl_xor(v, m, 32);
  return v;
}

// ---------------- prep: layernorm prototypes + l2-normalized f16 copy ----------------
__global__ void proto_prep(const float* __restrict__ protos,
                           const float* __restrict__ g, const float* __restrict__ b,
                           float* __restrict__ protosN, _Float16* __restrict__ phat) {
  int p = blockIdx.x * blockDim.x + threadIdx.x;
  if (p >= NPROT) return;
  const float* row = protos + (size_t)p * 16;
  float v[16], mean = 0.f;
#pragma unroll
  for (int k = 0; k < 16; k++) { v[k] = row[k]; mean += v[k]; }
  mean *= (1.f / 16.f);
  float var = 0.f;
#pragma unroll
  for (int k = 0; k < 16; k++) { float t = v[k] - mean; var += t * t; }
  var *= (1.f / 16.f);
  float inv = rsqrtf(var + 1e-5f);
  float n2 = 0.f;
#pragma unroll
  for (int k = 0; k < 16; k++) {
    float ln = (v[k] - mean) * inv * g[k] + b[k];
    protosN[(size_t)p * 16 + k] = ln;
    n2 += ln * ln;
  }
  float rn = 1.f / fmaxf(sqrtf(n2), 1e-12f);
#pragma unroll
  for (int k = 0; k < 16; k++)
    phat[(size_t)p * 16 + k] = (_Float16)(protosN[(size_t)p * 16 + k] * rn);
}

// ---------------- prep: proto_embed for all prototypes ----------------
__global__ void proto_embed_all(const float* __restrict__ protosN,
                                const float* __restrict__ projW, const float* __restrict__ projB,
                                float* __restrict__ PEall) {
  int idx = blockIdx.x * blockDim.x + threadIdx.x;  // 1024*128
  if (idx >= NPROT * 128) return;
  int p = idx >> 7, e = idx & 127;
  float s = projB[e];
#pragma unroll
  for (int k = 0; k < 16; k++) s += protosN[(size_t)p * 16 + k] * projW[(size_t)e * 16 + k];
  PEall[(size_t)p * 128 + e] = s;
}

// ---------------- prep: f32 -> f16 weight convert ----------------
__global__ void cvt_f16(const float* __restrict__ s, _Float16* __restrict__ d, int n) {
  for (int i = blockIdx.x * blockDim.x + threadIdx.x; i < n; i += gridDim.x * blockDim.x)
    d[i] = (_Float16)s[i];
}

// ---------------- zero pad buffer (address-select target for idle lanes) ----------------
__global__ void zero_buf(_Float16* __restrict__ z, int n) {
  int i = blockIdx.x * blockDim.x + threadIdx.x;
  if (i < n) z[i] = (_Float16)0.f;
}

// ---------------- init output with bias ----------------
__global__ void init_out(const float* __restrict__ outB, float* __restrict__ out) {
  int i = blockIdx.x * blockDim.x + threadIdx.x;  // 512*128
  if (i < 512 * 128) out[i] = outB[i & 127];
}

// ---------------- main fused kernel: one wave per 16-row tile ----------------
#define WAVES 4
__global__ __launch_bounds__(WAVES * 32) void fused_main(
    const float* __restrict__ x, const float* __restrict__ pg, const float* __restrict__ pb,
    const float* __restrict__ embB, const float* __restrict__ n1g, const float* __restrict__ n1b,
    const float* __restrict__ fb1, const float* __restrict__ fb2,
    const float* __restrict__ n2g, const float* __restrict__ n2b,
    const _Float16* __restrict__ phat, const float* __restrict__ PEall,
    const _Float16* __restrict__ embWh, const _Float16* __restrict__ W1h,
    const _Float16* __restrict__ W2h, const _Float16* __restrict__ zbuf,
    _Float16* __restrict__ zh) {
  __shared__ __align__(16) unsigned char smem[WAVES * 8192];
  const int wave = threadIdx.x >> 5;
  const int lane = threadIdx.x & 31;
  unsigned char* ws = smem + wave * 8192;
  const int tile = blockIdx.x * WAVES + wave;
  const int r0 = tile * 16;
  const int bidx = r0 >> 9;         // batch
  const int ndbase = r0 & 511;      // (n*32+d) of row m=0
  const int hi = lane >> 4, nn = lane & 15, kb = hi * 8;

  float* xbuf = (float*)ws;         // [16][16] layernormed
  float* xhbuf = xbuf + 256;        // [16][16] l2-normalized

  // ---- per-row layernorm over P=16 (lanes 0..15 each own one row) ----
  if (lane < 16) {
    int r = r0 + lane;
    int bb = r >> 9, rem = r & 511, n_ = rem >> 5, d_ = rem & 31;
    const float* xp = x + ((size_t)(bb * 16 + n_) * 16) * 32 + d_;
    float v[16], mean = 0.f;
#pragma unroll
    for (int p = 0; p < 16; p++) { v[p] = xp[(size_t)p * 32]; mean += v[p]; }
    mean *= (1.f / 16.f);
    float var = 0.f;
#pragma unroll
    for (int p = 0; p < 16; p++) { float t = v[p] - mean; var += t * t; }
    var *= (1.f / 16.f);
    float inv = rsqrtf(var + 1e-5f);
    float n2 = 0.f;
#pragma unroll
    for (int p = 0; p < 16; p++) {
      float ln = (v[p] - mean) * inv * pg[p] + pb[p];
      xbuf[lane * 16 + p] = ln; n2 += ln * ln;
    }
    float rn = 1.f / fmaxf(sqrtf(n2), 1e-12f);
#pragma unroll
    for (int p = 0; p < 16; p++) xhbuf[lane * 16 + p] = xbuf[lane * 16 + p] * rn;
  }
  __syncthreads();

  // ---- A fragments (16x32 f16, K=16 valid in lanes 0..15, lanes 16..31 zero) ----
  v16h a_xn = {}, a_xh = {};
#pragma unroll
  for (int e = 0; e < 16; e++) {
    a_xn[e] = hi ? (_Float16)0.f : (_Float16)xbuf[nn * 16 + e];
    a_xh[e] = hi ? (_Float16)0.f : (_Float16)xhbuf[nn * 16 + e];
  }
  __syncthreads();

  // ---- cosine sim vs 1024 prototypes + argmax ----
  // B fragment: lanes 0..15 hold K=0..15 (real data), lanes 16..31 hold K=16..31
  // which is the zero-padded region -> point them at a zero-filled buffer so the
  // load is exec-uniform (no divergent branch in the hot loop).
  const size_t pstep = (lane < 16) ? (size_t)256 : (size_t)0;  // halves per tile
  const _Float16* pbase = (lane < 16) ? (phat + (size_t)nn * 16) : zbuf;

  float bestv[8]; int besti[8];
#pragma unroll
  for (int v = 0; v < 8; v++) { bestv[v] = -3.4e38f; besti[v] = 0; }

  v16h bcur = {};
#pragma unroll
  for (int e = 0; e < 16; e++) bcur[e] = pbase[e];
  for (int t = 0; t < 64; t++) {
    v16h bnext = {};
    if (t < 63) {  // uniform branch; prefetch next tile's fragment
      const _Float16* pp = pbase + (size_t)(t + 1) * pstep;
#pragma unroll
      for (int e = 0; e < 16; e++) bnext[e] = pp[e];
    }
    v8f acc = {};
    acc = WMMA_F32_F16(a_xh, bcur, acc);
#pragma unroll
    for (int v = 0; v < 8; v++) {
      float val = acc[v];
      if (val > bestv[v]) { bestv[v] = val; besti[v] = t * 16 + nn; }
    }
    bcur = bnext;
  }
#pragma unroll
  for (int m = 1; m < 16; m <<= 1) {
#pragma unroll
    for (int v = 0; v < 8; v++) {
      float ov = __shfl_xor(bestv[v], m, 32);
      int   oi = __shfl_xor(besti[v], m, 32);
      if (ov > bestv[v] || (ov == bestv[v] && oi < besti[v])) { bestv[v] = ov; besti[v] = oi; }
    }
  }

  // ---- input_embed = xn @ embW^T + b  (8 WMMA, same address-select trick) ----
  const _Float16* ebase = (lane < 16) ? (embWh + (size_t)nn * 16) : zbuf;
  v8f ie[8];
  for (int t = 0; t < 8; t++) {
    const _Float16* wp = ebase + (size_t)t * pstep;
    v16h bf = {};
#pragma unroll
    for (int e = 0; e < 16; e++) bf[e] = wp[e];
    v8f acc = {};
    acc = WMMA_F32_F16(a_xn, bf, acc);
    float bias = embB[t * 16 + nn];
#pragma unroll
    for (int v = 0; v < 8; v++) acc[v] += bias;
    ie[t] = acc;
  }

  // ---- proto_embed: gather precomputed rows (L2-resident) ----
  v8f pe[8];
#pragma unroll
  for (int t = 0; t < 8; t++)
#pragma unroll
    for (int v = 0; v < 8; v++)
      pe[t][v] = PEall[(size_t)besti[v] * 128 + t * 16 + nn];

  // ---- row-wise: l2norms, softmax mix, residual, layernorm1 ----
  float nie[8], npe[8];
#pragma unroll
  for (int v = 0; v < 8; v++) {
    float a = 0.f, c = 0.f;
#pragma unroll
    for (int t = 0; t < 8; t++) { a += ie[t][v] * ie[t][v]; c += pe[t][v] * pe[t][v]; }
    nie[v] = 1.f / fmaxf(sqrtf(rowsum16(a)), 1e-12f);
    npe[v] = 1.f / fmaxf(sqrtf(rowsum16(c)), 1e-12f);
  }
  float w0[8], w1[8];
#pragma unroll
  for (int v = 0; v < 8; v++) {
    float a = 0.f, c = 0.f;
#pragma unroll
    for (int t = 0; t < 8; t++) {
      float q = ie[t][v] * nie[v], k2 = pe[t][v] * npe[v];
      a += q * q; c += q * k2;
    }
    float dkw = rowsum16(a), dkm = rowsum16(c);
    float mx = fmaxf(dkw, dkm);
    float e0 = __expf(dkw - mx), e1 = __expf(dkm - mx);
    float inv = 1.f / (e0 + e1);
    w0[v] = e0 * inv; w1[v] = e1 * inv;
  }
  float mean1[8], var1[8];
#pragma unroll
  for (int v = 0; v < 8; v++) {
    float s = 0.f;
#pragma unroll
    for (int t = 0; t < 8; t++) {
      float hp = ie[t][v] + w0[v] * ie[t][v] * nie[v] + w1[v] * pe[t][v] * npe[v];
      pe[t][v] = hp; s += hp;
    }
    mean1[v] = rowsum16(s) * (1.f / 128.f);
  }
#pragma unroll
  for (int v = 0; v < 8; v++) {
    float s = 0.f;
#pragma unroll
    for (int t = 0; t < 8; t++) { float d2 = pe[t][v] - mean1[v]; s += d2 * d2; }
    var1[v] = rowsum16(s) * (1.f / 128.f);
  }
  v8f h[8];
#pragma unroll
  for (int t = 0; t < 8; t++) {
    float g = n1g[t * 16 + nn], bb = n1b[t * 16 + nn];
#pragma unroll
    for (int v = 0; v < 8; v++)
      h[t][v] = (pe[t][v] - mean1[v]) * rsqrtf(var1[v] + 1e-5f) * g + bb;
  }

  // ---- stage h (D-layout -> LDS) and rebuild A fragments ----
  _Float16* hbuf = (_Float16*)ws;   // [16][128]
  __syncthreads();
#pragma unroll
  for (int t = 0; t < 8; t++)
#pragma unroll
    for (int v = 0; v < 8; v++)
      hbuf[(v + 8 * hi) * 128 + t * 16 + nn] = (_Float16)h[t][v];
  __syncthreads();
  v16h ah[4];
#pragma unroll
  for (int kc = 0; kc < 4; kc++) {
    v16h a = {};
#pragma unroll
    for (int e = 0; e < 8; e++) {
      a[e]     = hbuf[nn * 128 + kc * 32 + kb + e];
      a[e + 8] = hbuf[nn * 128 + kc * 32 + 16 + kb + e];
    }
    ah[kc] = a;
  }
  __syncthreads();

  // ---- FFN1: relu(h @ W1^T + b1) -> LDS f16 [16][256]  (64 WMMA) ----
  _Float16* abuf = (_Float16*)ws;
  for (int t = 0; t < 16; t++) {
    v8f acc = {};
#pragma unroll
    for (int kc = 0; kc < 4; kc++) {
      v16h bf = {};
      const _Float16* wp = W1h + (size_t)(t * 16 + nn) * 128 + kc * 32 + hi * 8;
#pragma unroll
      for (int e = 0; e < 16; e++) bf[e] = wp[(e & 7) + (e >> 3) * 16];
      acc = WMMA_F32_F16(ah[kc], bf, acc);
    }
    float bias = fb1[t * 16 + nn];
#pragma unroll
    for (int v = 0; v < 8; v++)
      abuf[(v + 8 * hi) * 256 + t * 16 + nn] = (_Float16)fmaxf(acc[v] + bias, 0.f);
  }
  __syncthreads();

  // ---- FFN2: @ W2^T  (64 WMMA) ----
  v8f f2[8];
  {
    v8f zeroc = {};
#pragma unroll
    for (int t = 0; t < 8; t++) f2[t] = zeroc;
  }
  for (int kc = 0; kc < 8; kc++) {
    v16h a = {};
#pragma unroll
    for (int e = 0; e < 8; e++) {
      a[e]     = abuf[nn * 256 + kc * 32 + kb + e];
      a[e + 8] = abuf[nn * 256 + kc * 32 + 16 + kb + e];
    }
#pragma unroll
    for (int t = 0; t < 8; t++) {
      v16h bf = {};
      const _Float16* wp = W2h + (size_t)(t * 16 + nn) * 256 + kc * 32 + hi * 8;
#pragma unroll
      for (int e = 0; e < 16; e++) bf[e] = wp[(e & 7) + (e >> 3) * 16];
      f2[t] = WMMA_F32_F16(a, bf, f2[t]);
    }
  }

  // ---- z = layernorm2(h + ffn2 + b2), store f16 to zh staging ----
#pragma unroll
  for (int t = 0; t < 8; t++) {
    float bias = fb2[t * 16 + nn];
#pragma unroll
    for (int v = 0; v < 8; v++) f2[t][v] += h[t][v] + bias;
  }
  float mean2[8], var2[8];
#pragma unroll
  for (int v = 0; v < 8; v++) {
    float s = 0.f;
#pragma unroll
    for (int t = 0; t < 8; t++) s += f2[t][v];
    mean2[v] = rowsum16(s) * (1.f / 128.f);
  }
#pragma unroll
  for (int v = 0; v < 8; v++) {
    float s = 0.f;
#pragma unroll
    for (int t = 0; t < 8; t++) { float d2 = f2[t][v] - mean2[v]; s += d2 * d2; }
    var2[v] = rowsum16(s) * (1.f / 128.f);
  }
#pragma unroll
  for (int t = 0; t < 8; t++) {
    float g = n2g[t * 16 + nn], bb = n2b[t * 16 + nn];
#pragma unroll
    for (int v = 0; v < 8; v++) {
      float zv = (f2[t][v] - mean2[v]) * rsqrtf(var2[v] + 1e-5f) * g + bb;
      int m = v + 8 * hi;
      zh[(size_t)bidx * KFIN + (size_t)(ndbase + m) * 128 + t * 16 + nn] = (_Float16)zv;
    }
  }
}

// ---------------- final GEMM: out += z_flat @ out_W^T, split-K + atomics ----------------
__global__ __launch_bounds__(256) void final_gemm(const _Float16* __restrict__ zh,
                                                  const _Float16* __restrict__ outWh,
                                                  float* __restrict__ out) {
  const int wave = threadIdx.x >> 5, lane = threadIdx.x & 31;
  const int wg = blockIdx.x * 8 + wave;
  const int tile = wg >> 5;           // 0..255 (32 M-tiles x 8 N-tiles)
  const int ks = wg & 31;             // split-K slice
  const int mt = tile >> 3, nt = tile & 7;
  const int hi = lane >> 4, nn = lane & 15, kb = hi * 8;
  const _Float16* ap = zh + (size_t)(mt * 16 + nn) * KFIN;
  const _Float16* bp = outWh + (size_t)(nt * 16 + nn) * KFIN;
  const int kcBeg = ks * 64, kcEnd = kcBeg + 64;

  v8f acc = {};
  // software pipeline: fetch next K-chunk's fragments while WMMA consumes current
  v16h ac = {}, bc = {};
  {
    int k0 = kcBeg * 32;
#pragma unroll
    for (int e = 0; e < 8; e++) { ac[e] = ap[k0 + kb + e]; ac[e + 8] = ap[k0 + 16 + kb + e]; }
#pragma unroll
    for (int e = 0; e < 16; e++) bc[e] = bp[k0 + kb + (e & 7) + (e >> 3) * 16];
  }
  for (int kc = kcBeg; kc < kcEnd; kc++) {
    v16h an = {}, bn = {};
    if (kc + 1 < kcEnd) {  // uniform branch
      int k0 = (kc + 1) * 32;
      __builtin_prefetch(bp + k0 + 1024, 0, 0);
      __builtin_prefetch(ap + k0 + 1024, 0, 0);
#pragma unroll
      for (int e = 0; e < 8; e++) { an[e] = ap[k0 + kb + e]; an[e + 8] = ap[k0 + 16 + kb + e]; }
#pragma unroll
      for (int e = 0; e < 16; e++) bn[e] = bp[k0 + kb + (e & 7) + (e >> 3) * 16];
    }
    acc = WMMA_F32_F16(ac, bc, acc);
    ac = an; bc = bn;
  }
#pragma unroll
  for (int v = 0; v < 8; v++) {
    int m = v + 8 * hi;
    atomicAdd(&out[(size_t)(mt * 16 + m) * 128 + nt * 16 + nn], acc[v]);
  }
}

// ---------------- host launcher ----------------
extern "C" void kernel_launch(void* const* d_in, const int* in_sizes, int n_in,
                              void* d_out, int out_size, void* d_ws, size_t ws_size,
                              hipStream_t stream) {
  const float* x        = (const float*)d_in[0];
  const float* protos   = (const float*)d_in[1];
  const float* patch_g  = (const float*)d_in[2];
  const float* patch_b  = (const float*)d_in[3];
  const float* proto_g  = (const float*)d_in[4];
  const float* proto_b  = (const float*)d_in[5];
  const float* emb_W    = (const float*)d_in[6];
  const float* emb_b    = (const float*)d_in[7];
  const float* proj_W   = (const float*)d_in[8];
  const float* proj_b   = (const float*)d_in[9];
  const float* ffn_W1   = (const float*)d_in[10];
  const float* ffn_b1   = (const float*)d_in[11];
  const float* ffn_W2   = (const float*)d_in[12];
  const float* ffn_b2   = (const float*)d_in[13];
  const float* n1_g     = (const float*)d_in[14];
  const float* n1_b     = (const float*)d_in[15];
  const float* n2_g     = (const float*)d_in[16];
  const float* n2_b     = (const float*)d_in[17];
  const float* out_W    = (const float*)d_in[18];
  const float* out_b    = (const float*)d_in[19];
  float* out = (float*)d_out;

  char* ws = (char*)d_ws;
  float*     protosN = (float*)(ws + 0);                       // 64 KB
  _Float16*  phat    = (_Float16*)(ws + (64 << 10));           // 32 KB
  float*     PEall   = (float*)(ws + (96 << 10));              // 512 KB
  _Float16*  embWh   = (_Float16*)(ws + (608 << 10));          // 4 KB
  _Float16*  W1h     = (_Float16*)(ws + (612 << 10));          // 64 KB
  _Float16*  W2h     = (_Float16*)(ws + (676 << 10));          // 64 KB
  _Float16*  zbuf    = (_Float16*)(ws + (740 << 10));          // 4 KB zeros
  _Float16*  outWh   = (_Float16*)(ws + (744 << 10));          // 16 MB
  _Float16*  zh      = (_Float16*)(ws + (744 << 10) + (size_t)128 * KFIN * 2); // 64 MB

  proto_prep<<<4, 256, 0, stream>>>(protos, proto_g, proto_b, protosN, phat);
  proto_embed_all<<<512, 256, 0, stream>>>(protosN, proj_W, proj_b, PEall);
  cvt_f16<<<8, 256, 0, stream>>>(emb_W, embWh, 128 * 16);
  cvt_f16<<<128, 256, 0, stream>>>(ffn_W1, W1h, 256 * 128);
  cvt_f16<<<128, 256, 0, stream>>>(ffn_W2, W2h, 128 * 256);
  cvt_f16<<<4096, 256, 0, stream>>>(out_W, outWh, 128 * KFIN);
  zero_buf<<<8, 256, 0, stream>>>(zbuf, 2048);
  init_out<<<256, 256, 0, stream>>>(out_b, out);
  fused_main<<<RROWS / 16 / WAVES, WAVES * 32, 0, stream>>>(
      x, patch_g, patch_b, emb_b, n1_g, n1_b, ffn_b1, ffn_b2, n2_g, n2_b,
      phat, PEall, embWh, W1h, W2h, zbuf, zh);
  final_gemm<<<1024, 256, 0, stream>>>(zh, outWh, out);
}
